// rcnn_71339406787086
// MI455X (gfx1250) — matrix operand order repaired and checked by reading, not compiled, not run on hardware
//
#include <hip/hip_runtime.h>
#include <hip/hip_bf16.h>
#include <math.h>

// Problem constants (match reference)
#define BB      8
#define NN      2000
#define CCLS    81
#define MAXI    100
#define MINCONF 0.05f
#define NMSTHR  0.3f
#define IMH     512
#define IMW     512
#define NPAD    2048   // sort padding (pow2 >= NN)
#define DPAD    128    // detection-dim padding for WMMA (4 x K=32)

typedef __attribute__((ext_vector_type(16))) _Float16 v16h;
typedef __attribute__((ext_vector_type(8)))  float    v8f;

// ---------------------------------------------------------------------------
// Kernel 1: per-proposal class argmax + box-delta refinement + sort-key build
// ---------------------------------------------------------------------------
__global__ void refine_kernel(const float* __restrict__ rois,
                              const float* __restrict__ probs,
                              const float* __restrict__ deltas,
                              const float* __restrict__ bstd,
                              float* __restrict__ scores,
                              float* __restrict__ boxes,
                              int*   __restrict__ cls,
                              unsigned long long* __restrict__ keys) {
    int idx = blockIdx.x * blockDim.x + threadIdx.x;
    if (idx >= BB * NN) return;

    // argmax over classes (first-max wins, like jnp.argmax)
    const float* p = probs + (size_t)idx * CCLS;
    int   best = 0;
    float bs   = p[0];
    for (int c = 1; c < CCLS; ++c) {
        float v = p[c];
        if (v > bs) { bs = v; best = c; }
    }

    const float* dl = deltas + ((size_t)idx * CCLS + best) * 4;
    float dy = dl[0] * bstd[0];
    float dx = dl[1] * bstd[1];
    float dh = dl[2] * bstd[2];
    float dw = dl[3] * bstd[3];

    const float* r = rois + (size_t)idx * 4;
    float y1 = r[0], x1 = r[1], y2 = r[2], x2 = r[3];
    float h = y2 - y1, w = x2 - x1;
    float cy = y1 + 0.5f * h + dy * h;
    float cx = x1 + 0.5f * w + dx * w;
    h = h * expf(dh);
    w = w * expf(dw);
    float ny1 = fminf(fmaxf(cy - 0.5f * h, 0.f), 1.f);
    float nx1 = fminf(fmaxf(cx - 0.5f * w, 0.f), 1.f);
    float ny2 = fminf(fmaxf(cy + 0.5f * h, 0.f), 1.f);
    float nx2 = fminf(fmaxf(cx + 0.5f * w, 0.f), 1.f);

    scores[idx]      = bs;
    boxes[idx*4 + 0] = ny1;
    boxes[idx*4 + 1] = nx1;
    boxes[idx*4 + 2] = ny2;
    boxes[idx*4 + 3] = nx2;
    cls[idx]         = best;

    bool valid = (best > 0) && (bs >= MINCONF);
    float keyf = valid ? bs : -INFINITY;
    unsigned u = __float_as_uint(keyf);
    u = (u & 0x80000000u) ? ~u : (u | 0x80000000u);   // order-preserving bits
    unsigned n = (unsigned)(idx % NN);
    // descending sort on key => score desc, lower index first on ties (stable)
    keys[idx] = ((unsigned long long)u << 32) | (0xFFFFFFFFu - n);
}

// ---------------------------------------------------------------------------
// Kernel 2: per-batch bitonic sort (descending) of composite keys in LDS
// ---------------------------------------------------------------------------
__global__ void sort_kernel(const unsigned long long* __restrict__ keys,
                            int* __restrict__ order) {
    __shared__ unsigned long long k[NPAD];     // 16 KB
    int b = blockIdx.x;

    for (int i = threadIdx.x; i < NPAD; i += blockDim.x) {
        if (i < NN) {
            k[i] = keys[b * NN + i];
        } else {
            // -inf score, index beyond NN: sorts after every real entry
            k[i] = (0x007FFFFFull << 32) | (unsigned long long)(0xFFFFFFFFu - (unsigned)i);
        }
    }

    for (int kk = 2; kk <= NPAD; kk <<= 1) {
        for (int j = kk >> 1; j > 0; j >>= 1) {
            __syncthreads();
            for (int i = threadIdx.x; i < NPAD; i += blockDim.x) {
                int ixj = i ^ j;
                if (ixj > i) {
                    unsigned long long a = k[i], c = k[ixj];
                    bool up = ((i & kk) == 0);          // descending overall
                    bool sw = up ? (a < c) : (a > c);
                    if (sw) { k[i] = c; k[ixj] = a; }
                }
            }
        }
    }
    __syncthreads();
    for (int i = threadIdx.x; i < NN; i += blockDim.x)
        order[b * NN + i] = (int)(0xFFFFFFFFu - (unsigned)(k[i] & 0xFFFFFFFFull));
}

// ---------------------------------------------------------------------------
// Kernel 3: greedy NMS (per-class via class match == class-offset trick) and
//           inline emission of det rows (top-100 kept == first 100 kept,
//           since candidates are score-sorted). One workgroup per batch.
// ---------------------------------------------------------------------------
__device__ __forceinline__ float iou_pair(float ay1, float ax1, float ay2, float ax2,
                                          float by1, float bx1, float by2, float bx2) {
    float areaA = (ay2 - ay1) * (ax2 - ax1);
    float areaB = (by2 - by1) * (bx2 - bx1);
    float yy1 = fmaxf(ay1, by1), xx1 = fmaxf(ax1, bx1);
    float yy2 = fminf(ay2, by2), xx2 = fminf(ax2, bx2);
    float inter = fmaxf(yy2 - yy1, 0.f) * fmaxf(xx2 - xx1, 0.f);
    float uni   = areaA + areaB - inter;
    return inter / fmaxf(uni, 1e-12f);
}

__global__ void nms_kernel(const int*   __restrict__ order,
                           const float* __restrict__ boxes,
                           const int*   __restrict__ cls,
                           const float* __restrict__ scores,
                           float* __restrict__ det) {
    __shared__ float sy1[NN], sx1[NN], sy2[NN], sx2[NN];   // 32 KB
    __shared__ float sscore[NN];                           // 8 KB
    __shared__ short scls[NN];                             // 4 KB
    __shared__ unsigned char skeep[NN], svalid[NN];        // 4 KB
    __shared__ int counts[CCLS];
    __shared__ int sflag;
    __shared__ int sndet;

    int b = blockIdx.x;
    int t = threadIdx.x;

    for (int i = t; i < NN; i += blockDim.x) {
        int n    = order[b * NN + i];
        int g    = b * NN + n;
        float a0 = boxes[g*4+0], a1 = boxes[g*4+1], a2 = boxes[g*4+2], a3 = boxes[g*4+3];
        int   c  = cls[g];
        float s  = scores[g];
        sy1[i] = a0; sx1[i] = a1; sy2[i] = a2; sx2[i] = a3;
        sscore[i] = s;
        scls[i]   = (short)c;
        skeep[i]  = 0;
        svalid[i] = (c > 0 && s >= MINCONF) ? 1 : 0;
    }
    for (int i = t; i < CCLS; i += blockDim.x) counts[i] = 0;
    if (t == 0) sndet = 0;

    for (int i = 0; i < NN; ++i) {
        if (t == 0) sflag = 0;
        __syncthreads();
        int   ci = scls[i];
        float a0 = sy1[i], a1 = sx1[i], a2 = sy2[i], a3 = sx2[i];
        for (int j = t; j < i; j += blockDim.x) {
            if (skeep[j] && scls[j] == ci) {
                if (iou_pair(sy1[j], sx1[j], sy2[j], sx2[j], a0, a1, a2, a3) > NMSTHR)
                    sflag = 1;                 // benign LDS race: all write 1
            }
        }
        __syncthreads();
        if (t == 0) {
            bool ok = svalid[i] && (sflag == 0) && (counts[ci] < MAXI);
            skeep[i] = ok ? 1 : 0;
            if (ok) {
                counts[ci] += 1;
                if (sndet < MAXI) {
                    float* row = det + ((size_t)b * MAXI + sndet) * 6;
                    row[0] = a0; row[1] = a1; row[2] = a2; row[3] = a3;
                    row[4] = (float)ci;
                    row[5] = sscore[i];
                    sndet += 1;
                }
            }
        }
        // next iteration's flag-reset + barrier publishes skeep[i]/counts/sndet
    }
    __syncthreads();
    int nd = sndet;
    for (int rr = t; rr < MAXI; rr += blockDim.x) {
        if (rr >= nd) {
            float* row = det + ((size_t)b * MAXI + rr) * 6;
            row[0] = 0.f; row[1] = 0.f; row[2] = 0.f;
            row[3] = 0.f; row[4] = 0.f; row[5] = 0.f;
        }
    }
}

// ---------------------------------------------------------------------------
// Kernel 4: attention mask via WMMA. mask[b,h,w] = (sum_d iny[d,h]*inx[d,w]) > 0
// Per-batch GEMM (512 x 512 x 128) with f16 0/1 indicator fragments,
// accumulated by v_wmma_f32_16x16x32_f16. One wave per 16x16 output tile.
// Fragment build is branchless: bitwise & of two v_cmp results + v_cndmask,
// box extents staged in LDS as float2 pairs so each element is one ds_load_b64.
// ---------------------------------------------------------------------------
__global__ void mask_kernel(const float* __restrict__ det,
                            float* __restrict__ mask) {
    __shared__ float2 sy[DPAD];   // {y1, y2} per detection slot
    __shared__ float2 sx[DPAD];   // {x1, x2} per detection slot

    int b    = blockIdx.x >> 7;        // 128 blocks per batch
    int tblk = blockIdx.x & 127;
    int t    = threadIdx.x;

    if (t < DPAD) {
        float a = 0.f, bb = 0.f, cc = 0.f, dd = 0.f;
        if (t < MAXI) {
            const float* row = det + ((size_t)b * MAXI + t) * 6;
            a = row[0]; bb = row[1]; cc = row[2]; dd = row[3];
        }
        // zero boxes (padding / invalid rows) cover no pixel: y>=0 && y<0 == false
        sy[t] = make_float2(a, cc);
        sx[t] = make_float2(bb, dd);
    }
    __syncthreads();

    int wid  = t >> 5;                 // wave in block: 0..7
    int lane = t & 31;
    int tile = tblk * 8 + wid;         // 0..1023 within batch
    int th   = tile >> 5, tw = tile & 31;
    int hb   = th * 16,  wb = tw * 16;
    int m    = lane & 15;
    int hi   = lane >> 4;

    float y = ((float)(hb + m) + 0.5f) * (1.0f / (float)IMH);
    float x = ((float)(wb + m) + 0.5f) * (1.0f / (float)IMW);

    const _Float16 one  = (_Float16)1.0f;
    const _Float16 zero = (_Float16)0.0f;

    v8f acc = {};
    #pragma unroll
    for (int kc = 0; kc < 4; ++kc) {
        v16h a, bf;
        #pragma unroll
        for (int r = 0; r < 8; ++r) {
            // A (16x32 f16): VGPR0-3 -> K 0..7 (lane-lo) / 8..15 (lane-hi),
            //                VGPR4-7 -> K 16..23 / 24..31
            int ka = (r < 4) ? (2*r + hi*8) : (2*r + 8 + hi*8);
            int da = kc * 32 + ka;
            float2 p0 = sy[da];
            float2 p1 = sy[da + 1];
            // bitwise & (no short-circuit) -> branchless v_cmp/v_cmp/and/cndmask
            bool ta0 = (y >= p0.x) & (y < p0.y);
            bool ta1 = (y >= p1.x) & (y < p1.y);
            a[2*r]   = ta0 ? one : zero;
            a[2*r+1] = ta1 ? one : zero;

            // B (32x16 f16): lanes = N, same K order as A: K 0..7/16..23 (lo), 8..15/24..31 (hi)
            int kb = (r < 4) ? (2*r + hi*8) : (2*r + 8 + hi*8);
            int db = kc * 32 + kb;
            float2 q0 = sx[db];
            float2 q1 = sx[db + 1];
            bool tb0 = (x >= q0.x) & (x < q0.y);
            bool tb1 = (x >= q1.x) & (x < q1.y);
            bf[2*r]   = tb0 ? one : zero;
            bf[2*r+1] = tb1 ? one : zero;
        }
        acc = __builtin_amdgcn_wmma_f32_16x16x32_f16(
                  /*neg_a=*/false, a, /*neg_b=*/false, bf,
                  /*c_mod=*/(short)0, acc, /*reuse_a=*/false, /*reuse_b=*/false);
    }

    size_t base = (size_t)b * IMH * IMW;
    #pragma unroll
    for (int r = 0; r < 8; ++r) {
        int row = hb + r + hi * 8;     // C/D layout: VGPR r -> M=r (lo) / r+8 (hi)
        int col = wb + m;
        mask[base + (size_t)row * IMW + col] = (acc[r] > 0.f) ? 1.0f : 0.0f;
    }
}

// ---------------------------------------------------------------------------
// Host-side launcher
// ---------------------------------------------------------------------------
extern "C" void kernel_launch(void* const* d_in, const int* in_sizes, int n_in,
                              void* d_out, int out_size, void* d_ws, size_t ws_size,
                              hipStream_t stream) {
    (void)in_sizes; (void)n_in; (void)out_size; (void)ws_size;

    const float* rois   = (const float*)d_in[0];   // [B,N,4]
    const float* probs  = (const float*)d_in[1];   // [B,N,C]
    const float* deltas = (const float*)d_in[2];   // [B,N,C,4]
    const float* bstd   = (const float*)d_in[3];   // [4]

    float* det  = (float*)d_out;                   // [B,100,6]
    float* mask = (float*)d_out + (size_t)BB * MAXI * 6;  // [B,512,512]

    // Workspace layout (~576 KB; offsets 8B-aligned by construction)
    float* scores = (float*)d_ws;                              // B*N
    float* boxes  = scores + (size_t)BB * NN;                  // B*N*4
    int*   cls    = (int*)(boxes + (size_t)BB * NN * 4);       // B*N
    unsigned long long* keys = (unsigned long long*)(cls + (size_t)BB * NN);  // B*N
    int*   order  = (int*)(keys + (size_t)BB * NN);            // B*N

    refine_kernel<<<(BB * NN + 255) / 256, 256, 0, stream>>>(
        rois, probs, deltas, bstd, scores, boxes, cls, keys);

    sort_kernel<<<BB, 1024, 0, stream>>>(keys, order);

    nms_kernel<<<BB, 1024, 0, stream>>>(order, boxes, cls, scores, det);

    mask_kernel<<<BB * 128, 256, 0, stream>>>(det, mask);
}
